// Decoder_70222715289879
// MI455X (gfx1250) — matrix operand order repaired and checked by reading, not compiled, not run on hardware
//
#include <hip/hip_runtime.h>
#include <math.h>

// ---- problem dims ----
#define HIDN 256
#define H2   512
#define EMBD 128
#define BB   16
#define KKk  4
#define NRr  5
#define NQq  5
#define LRr  100
#define LQq  60
#define LLl  460            // LQ + K*LR
#define VOCB 50000
#define VOUT 50100
#define EPSF 1e-12f

typedef __attribute__((ext_vector_type(16))) _Float16 v16h;
typedef __attribute__((ext_vector_type(8)))  float    v8f;

// ---- output offsets (floats) ----
#define OUT_FD  0
#define OUT_H1  801600
#define OUT_C1  805696
#define OUT_CTT 809792
#define OUT_AQR 817984
#define OUT_COV 825344

// ---- workspace offsets (floats) ----
#define OFF_ST   ((size_t)0)
#define OFF_X    (OFF_ST + 8192)
#define OFF_TQ   (OFF_X + 2048)
#define OFF_WNQ  (OFF_TQ + 2048)
#define OFF_QAT  (OFF_WNQ + 128)
#define OFF_TA   (OFF_QAT + 2048)
#define OFF_TO   (OFF_TA + 2048)
#define OFF_WNA  (OFF_TO + 2048)
#define OFF_WNO  (OFF_WNA + 320)
#define OFF_AQ   (OFF_WNO + 320)
#define OFF_AR   (OFF_AQ + 3840)
#define OFF_E    (OFF_AR + 6400)
#define OFF_A    (OFF_E + 7360)
#define OFF_CT   (OFF_A + 7360)
#define OFF_HST  (OFF_CT + 8192)
#define OFF_PG   (OFF_HST + 16384)
#define OFF_VM   (OFF_PG + 16)
#define OFF_VS   (OFF_VM + 16)
#define OFF_HID  (OFF_VS + 16)
#define OFF_SW   (OFF_HID + 8192)
#define OFF_QT   (OFF_SW + 8192)
#define OFF_QU   (OFF_QT + 491520)
#define OFF_U    (OFF_QU + 491520)
#define OFF_RT   (OFF_U + 384000)
#define OFF_PAI  (OFF_RT + 3276800)
#define OFF_PW   (OFF_PAI + 3768320)
#define OFF_LOG  (OFF_QT)   // alias: q_t/qU/U dead before logits are computed

#if defined(__AMDGCN__) && __has_builtin(__builtin_amdgcn_s_wait_asynccnt)
#define HAVE_ASYNC_LDS 1
#endif

__device__ __forceinline__ float sigm(float x){ return 1.0f/(1.0f+expf(-x)); }

// ---------------- WMMA helpers (A/B k order: contiguous k = 16*(lane/16) + s in both) ----------------
__device__ __forceinline__ v8f wmma16(v16h a, v16h b, v8f c){
  return __builtin_amdgcn_wmma_f32_16x16x32_f16(false, a, false, b, (short)0, c, false, false);
}

// A: row-major f32 [M x lda], tile (m0,k0), rows >= mlim are zero
__device__ __forceinline__ v16h load_a16(const float* A, int lda, int m0, int k0, int mlim, int lane){
  v16h a;
  int m  = m0 + (lane & 15);
  int hi = (lane & 16) ? 16 : 0;
  if (m < mlim) {
    const float* p = A + (size_t)m * lda + k0 + hi;
#pragma unroll
    for (int i = 0; i < 8; ++i) { a[i] = (_Float16)p[i]; a[8+i] = (_Float16)p[8+i]; }
  } else {
#pragma unroll
    for (int i = 0; i < 16; ++i) a[i] = (_Float16)0.0f;
  }
  return a;
}

// B given as row-major Bt [N x ldb] so that B[k][n] = Bt[n*ldb + k]; cols >= nlim zero
__device__ __forceinline__ v16h load_bt16(const float* Bt, int ldb, int n0, int k0, int nlim, int lane){
  v16h b;
  int n  = n0 + (lane & 15);
  int ko = k0 + ((lane & 16) ? 16 : 0);
  if (n < nlim) {
    const float* p = Bt + (size_t)n * ldb + ko;
#pragma unroll
    for (int i = 0; i < 16; ++i) b[i] = (_Float16)p[i];
  } else {
#pragma unroll
    for (int i = 0; i < 16; ++i) b[i] = (_Float16)0.0f;
  }
  return b;
}

#ifdef HAVE_ASYNC_LDS
// Async-copy one 16x32 f32 K-chunk (2 KB) of a row-major [16 x ldk] panel into LDS.
// 4 x async b128 per wave: 32 lanes x 16 B each; 8 lanes cover one row's 32 floats.
// Inline asm per CDNA5 ISA VGLOBAL: VDST = LDS byte address VGPR,
// VADDR = 64-bit global address pair, no SADDR ("off"). Tracked by ASYNCcnt.
__device__ __forceinline__ void async_copy_chunk(const float* gsrc, int ldk, float* lds, int lane){
#pragma unroll
  for (int i = 0; i < 4; ++i) {
    int elem = i*128 + lane*4;          // float index in chunk
    int row  = elem >> 5;
    int kk   = elem & 31;
    unsigned long long gaddr = (unsigned long long)(size_t)(gsrc + (size_t)row*ldk + kk);
    unsigned ldsoff = (unsigned)(size_t)(lds + elem);  // low 32 bits of generic LDS ptr = LDS offset
    asm volatile("global_load_async_to_lds_b128 %0, %1, off"
                 :: "v"(ldsoff), "v"(gaddr)
                 : "memory");
  }
}
// B fragment from LDS chunk laid out [16 rows][32 k]
__device__ __forceinline__ v16h load_b_lds(const float* lds32, int lane){
  v16h b;
  int n  = lane & 15;
  int ko = (lane & 16) ? 16 : 0;
  const float* p = lds32 + n*32 + ko;
#pragma unroll
  for (int i = 0; i < 16; ++i) b[i] = (_Float16)p[i];
  return b;
}
#endif

// ---------------- tiny scalar kernels ----------------
__global__ void k_x(const float* y_t, const float* c_t0, const float* xc_w, const float* xc_b, float* X){
  int idx = blockIdx.x*blockDim.x + threadIdx.x; if (idx >= BB*EMBD) return;
  int b = idx / EMBD, e = idx % EMBD;
  float acc = xc_b[e];
  const float* w = xc_w + (size_t)e*(EMBD+H2);
  for (int j = 0; j < EMBD; ++j) acc += y_t[b*EMBD+j]*w[j];
  for (int j = 0; j < H2;   ++j) acc += c_t0[b*H2+j]*w[EMBD+j];
  X[idx] = acc;
}

__global__ void k_lstm(const float* X, const float* h0, const float* c0,
                       const float* w_ih, const float* w_hh, const float* b_ih, const float* b_hh,
                       float* ST, float* out){
  int idx = blockIdx.x*blockDim.x + threadIdx.x; if (idx >= BB*HIDN) return;
  int b = idx / HIDN, u = idx % HIDN;
  float g4[4];
#pragma unroll
  for (int gi = 0; gi < 4; ++gi) {
    int row = gi*HIDN + u;
    float acc = b_ih[row] + b_hh[row];
    const float* wi = w_ih + (size_t)row*EMBD;
    const float* wh = w_hh + (size_t)row*HIDN;
    for (int j = 0; j < EMBD; ++j) acc += X[b*EMBD+j]*wi[j];
    for (int j = 0; j < HIDN; ++j) acc += h0[b*HIDN+j]*wh[j];
    g4[gi] = acc;
  }
  float c1 = sigm(g4[1])*c0[idx] + sigm(g4[0])*tanhf(g4[2]);
  float h1 = sigm(g4[3])*tanhf(c1);
  out[OUT_H1 + idx] = h1;
  out[OUT_C1 + idx] = c1;
  ST[(size_t)b*H2 + u]        = h1;
  ST[(size_t)b*H2 + HIDN + u] = c1;
}

__global__ void k_tq(const float* ST, const float* W, float* T){
  int idx = blockIdx.x*blockDim.x + threadIdx.x; if (idx >= BB*EMBD) return;
  int b = idx/EMBD, e = idx%EMBD;
  float acc = 0.f;
  for (int d = 0; d < H2; ++d) acc += ST[(size_t)b*H2+d]*W[(size_t)d*EMBD+e];
  T[idx] = acc;
}
__global__ void k_ta(const float* ST, const float* QAT, const float* W, float* T){
  int idx = blockIdx.x*blockDim.x + threadIdx.x; if (idx >= BB*EMBD) return;
  int b = idx/EMBD, e = idx%EMBD;
  float acc = 0.f;
  for (int d = 0; d < H2;   ++d) acc += ST[(size_t)b*H2+d]   *W[(size_t)d*EMBD+e];
  for (int d = 0; d < EMBD; ++d) acc += QAT[(size_t)b*EMBD+d]*W[(size_t)(H2+d)*EMBD+e];
  T[idx] = acc;
}
__global__ void k_to(const float* ST, const float* o_m, const float* W, float* T){
  int idx = blockIdx.x*blockDim.x + threadIdx.x; if (idx >= BB*EMBD) return;
  int b = idx/EMBD, e = idx%EMBD;
  float acc = 0.f;
  for (int d = 0; d < H2;      ++d) acc += ST[(size_t)b*H2+d]         *W[(size_t)d*EMBD+e];
  for (int d = 0; d < H2+EMBD; ++d) acc += o_m[(size_t)b*(H2+EMBD)+d]*W[(size_t)(H2+d)*EMBD+e];
  T[idx] = acc;
}

__global__ void k_wq(const float* T, const float* Aemb, float* WN){
  int b = blockIdx.x, t = threadIdx.x;
  __shared__ float wv[NQq]; __shared__ float ssum;
  if (t < NQq) {
    float acc = 0.f;
    const float* a = Aemb + (size_t)(b*NQq + t)*EMBD;
    for (int e = 0; e < EMBD; ++e) acc += T[(size_t)b*EMBD+e]*a[e];
    wv[t] = tanhf(acc);
  }
  __syncthreads();
  if (t == 0) { float s = 0.f; for (int n = 0; n < NQq; ++n) s += wv[n]; ssum = s + EPSF; }
  __syncthreads();
  if (t < NQq) WN[b*NQq + t] = wv[t]/ssum;
}
__global__ void k_wr(const float* T, const float* Aemb, float* WN){
  int bk = blockIdx.x, t = threadIdx.x;
  int b = bk / KKk;
  __shared__ float wv[NRr]; __shared__ float ssum;
  if (t < NRr) {
    float acc = 0.f;
    const float* a = Aemb + (size_t)(bk*NRr + t)*EMBD;
    for (int e = 0; e < EMBD; ++e) acc += T[(size_t)b*EMBD+e]*a[e];
    wv[t] = tanhf(acc);
  }
  __syncthreads();
  if (t == 0) { float s = 0.f; for (int n = 0; n < NRr; ++n) s += wv[n]; ssum = s + EPSF; }
  __syncthreads();
  if (t < NRr) WN[bk*NRr + t] = wv[t]/ssum;
}

__global__ void k_qt(const float* WN, const float* h_q, float* QT){
  int idx = blockIdx.x*blockDim.x + threadIdx.x; if (idx >= BB*LQq*H2) return;
  int b = idx/(LQq*H2); int r = idx%(LQq*H2); int l = r/H2; int h = r%H2;
  float acc = 0.f;
#pragma unroll
  for (int n = 0; n < NQq; ++n)
    acc += WN[b*NQq+n]*h_q[((size_t)(b*NQq+n)*LQq + l)*H2 + h];
  QT[idx] = acc;
}
__global__ void k_qat(const float* WN, const float* q_a, float* QAT){
  int idx = blockIdx.x*blockDim.x + threadIdx.x; if (idx >= BB*EMBD) return;
  int b = idx/EMBD, e = idx%EMBD;
  float acc = 0.f;
#pragma unroll
  for (int n = 0; n < NQq; ++n) acc += WN[b*NQq+n]*q_a[(size_t)(b*NQq+n)*EMBD+e];
  QAT[idx] = acc;
}
// r_t = sum_n (wn_a+wn_o)[b,k,n] * h_r[b,k,n,l,h]  (single pass over h_r)
__global__ void k_rt(const float* WNA, const float* WNO, const float* h_r, float* RT){
  int idx = blockIdx.x*blockDim.x + threadIdx.x; if (idx >= BB*KKk*LRr*H2) return;
  int bk = idx/(LRr*H2); int r = idx%(LRr*H2); int l = r/H2; int h = r%H2;
  float acc = 0.f;
#pragma unroll
  for (int n = 0; n < NRr; ++n)
    acc += (WNA[bk*NRr+n]+WNO[bk*NRr+n]) * h_r[((size_t)(bk*NRr+n)*LRr + l)*H2 + h];
  RT[idx] = acc;
}

// ---------------- WMMA GEMMs ----------------
// qU(960x512) = QT(960x512) x U_w^T
__global__ __launch_bounds__(32) void k_gemm_qU(const float* QT, const float* Uw, float* QU){
  int blk = blockIdx.x; int mt = blk >> 5, nt = blk & 31; int lane = threadIdx.x;
  const float* arow = QT + (size_t)(mt*16 + (lane & 15))*H2;
  v8f c = {};
  for (int k0 = 0; k0 < H2; k0 += 32) {
    __builtin_prefetch(arow + k0 + 64, 0, 0);     // global_prefetch_b8: A chunk two steps ahead
    v16h a = load_a16(QT, H2, mt*16, k0, BB*LQq, lane);
    v16h b = load_bt16(Uw, H2, nt*16, k0, H2, lane);
    c = wmma16(a, b, c);
  }
  int n = nt*16 + (lane & 15);
  int mb = mt*16 + ((lane & 16) ? 8 : 0);
#pragma unroll
  for (int j = 0; j < 8; ++j) QU[(size_t)(mb+j)*H2 + n] = c[j];
}
// U[bk,q,r] = tanh( qU[b] (60x512) x rt[bk]^T (512x100) )
__global__ __launch_bounds__(32) void k_gemm_U(const float* QU, const float* RT, float* U){
  int blk = blockIdx.x;
  int bk = blk / 28; int t = blk % 28; int mt = t / 7, nt = t % 7;
  int lane = threadIdx.x;
  const float* A = QU + (size_t)(bk/KKk)*LQq*H2;
  const float* B = RT + (size_t)bk*LRr*H2;
  v8f c = {};
  for (int k0 = 0; k0 < H2; k0 += 32) {
    v16h a = load_a16(A, H2, mt*16, k0, LQq, lane);
    v16h b = load_bt16(B, H2, nt*16, k0, LRr, lane);
    c = wmma16(a, b, c);
  }
  int n = nt*16 + (lane & 15);
  int mb = mt*16 + ((lane & 16) ? 8 : 0);
  if (n < LRr) {
#pragma unroll
    for (int j = 0; j < 8; ++j) {
      int m = mb + j;
      if (m < LQq) U[(size_t)bk*LQq*LRr + (size_t)m*LRr + n] = tanhf(c[j]);
    }
  }
}
// pW(7360x512) = PAI(7360x512) x Wqr^T
__global__ __launch_bounds__(32) void k_gemm_pW(const float* PAI, const float* Wqr, float* PW){
  int blk = blockIdx.x; int mt = blk >> 5, nt = blk & 31; int lane = threadIdx.x;
  const float* arow = PAI + (size_t)(mt*16 + (lane & 15))*H2;
  v8f c = {};
  for (int k0 = 0; k0 < H2; k0 += 32) {
    __builtin_prefetch(arow + k0 + 64, 0, 0);     // global_prefetch_b8: A chunk two steps ahead
    v16h a = load_a16(PAI, H2, mt*16, k0, BB*LLl, lane);
    v16h b = load_bt16(Wqr, H2, nt*16, k0, H2, lane);
    c = wmma16(a, b, c);
  }
  int n = nt*16 + (lane & 15);
  int mb = mt*16 + ((lane & 16) ? 8 : 0);
#pragma unroll
  for (int j = 0; j < 8; ++j) PW[(size_t)(mb+j)*H2 + n] = c[j];
}

// logits(16x50000) = HID(16x512) x pv2_w^T + pv2_b ; one M tile, 3125 N tiles.
// pv2_w is the dominant HBM stream (102 MB): double-buffer 16x32 K-chunks of the
// B panel through LDS with async global->LDS copies so the next chunk's HBM fetch
// overlaps the current chunk's cvt+WMMA (ASYNCcnt producer/consumer pattern).
#ifdef HAVE_ASYNC_LDS
__global__ __launch_bounds__(32) void k_gemm_logits(const float* HID, const float* pv2w,
                                                    const float* pv2b, float* LOG){
  __shared__ float lbuf[2][512];
  int nt = blockIdx.x; int lane = threadIdx.x;
  const float* Bbase = pv2w + (size_t)(nt*16)*H2;
  async_copy_chunk(Bbase, H2, &lbuf[0][0], lane);            // chunk 0 in flight
  v8f c = {};
  for (int kc = 0; kc < 15; ++kc) {
    async_copy_chunk(Bbase + (kc+1)*32, H2, &lbuf[(kc+1)&1][0], lane);  // next chunk
    __builtin_amdgcn_s_wait_asynccnt(4);                      // chunk kc landed in LDS
    v16h a = load_a16(HID, H2, 0, kc*32, BB, lane);
    v16h b = load_b_lds(&lbuf[kc&1][0], lane);
    c = wmma16(a, b, c);
  }
  __builtin_amdgcn_s_wait_asynccnt(0);                        // last chunk
  {
    v16h a = load_a16(HID, H2, 0, 15*32, BB, lane);
    v16h b = load_b_lds(&lbuf[1][0], lane);
    c = wmma16(a, b, c);
  }
  int n = nt*16 + (lane & 15);
  int mb = (lane & 16) ? 8 : 0;
  float bias = pv2b[n];
#pragma unroll
  for (int j = 0; j < 8; ++j) LOG[(size_t)(mb+j)*VOCB + n] = c[j] + bias;
}
#else
__global__ __launch_bounds__(32) void k_gemm_logits(const float* HID, const float* pv2w,
                                                    const float* pv2b, float* LOG){
  int nt = blockIdx.x; int lane = threadIdx.x;
  v8f c = {};
  for (int k0 = 0; k0 < H2; k0 += 32) {
    v16h a = load_a16(HID, H2, 0, k0, BB, lane);
    v16h b = load_bt16(pv2w, H2, nt*16, k0, VOCB, lane);
    c = wmma16(a, b, c);
  }
  int n = nt*16 + (lane & 15);
  int mb = (lane & 16) ? 8 : 0;
  float bias = pv2b[n];
#pragma unroll
  for (int j = 0; j < 8; ++j) LOG[(size_t)(mb+j)*VOCB + n] = c[j] + bias;
}
#endif

// ---------------- attention epilogues ----------------
__global__ void k_alpha(const float* U, float* AQ, float* AR){
  int bk = blockIdx.x, t = threadIdx.x;
  const float* u = U + (size_t)bk*LQq*LRr;
  __shared__ float s[128]; __shared__ float gm, gs;
  float mq = -__builtin_inff();
  if (t < LQq) { for (int r = 0; r < LRr; ++r) mq = fmaxf(mq, u[(size_t)t*LRr+r]); }
  s[t] = (t < LQq) ? mq : -__builtin_inff();
  __syncthreads();
  if (t == 0) { float m = -__builtin_inff(); for (int q = 0; q < LQq; ++q) m = fmaxf(m, s[q]); gm = m; }
  __syncthreads();
  float ex = (t < LQq) ? expf(s[t]-gm) : 0.f;
  __syncthreads(); s[t] = ex; __syncthreads();
  if (t == 0) { float sum = 0.f; for (int q = 0; q < LQq; ++q) sum += s[q]; gs = sum; }
  __syncthreads();
  if (t < LQq) AQ[bk*LQq + t] = ex/gs;
  __syncthreads();
  float mr = -__builtin_inff();
  if (t < LRr) { for (int q = 0; q < LQq; ++q) mr = fmaxf(mr, u[(size_t)q*LRr+t]); }
  s[t] = (t < LRr) ? mr : -__builtin_inff();
  __syncthreads();
  if (t == 0) { float m = -__builtin_inff(); for (int r = 0; r < LRr; ++r) m = fmaxf(m, s[r]); gm = m; }
  __syncthreads();
  float ex2 = (t < LRr) ? expf(s[t]-gm) : 0.f;
  __syncthreads(); s[t] = ex2; __syncthreads();
  if (t == 0) { float sum = 0.f; for (int r = 0; r < LRr; ++r) sum += s[r]; gs = sum; }
  __syncthreads();
  if (t < LRr) AR[bk*LRr + t] = ex2/gs;
}

__global__ void k_pai(const float* AQ, const float* AR, const float* q_mask, const float* r_mask,
                      const float* QT, const float* RT, float* PAI){
  int idx = blockIdx.x*blockDim.x + threadIdx.x; if (idx >= BB*LLl*H2) return;
  int b = idx/(LLl*H2); int r = idx%(LLl*H2); int l = r/H2; int h = r%H2;
  float v;
  if (l < LQq) {
    float aks = 0.f;
#pragma unroll
    for (int k = 0; k < KKk; ++k) aks += AQ[(b*KKk+k)*LQq + l];
    v = aks * q_mask[b*LQq+l] * QT[((size_t)b*LQq+l)*H2 + h] * 0.25f;
  } else {
    int li = l - LQq; int k = li/LRr; int rr = li%LRr; int bkr = (b*KKk+k)*LRr + rr;
    v = AR[bkr]*r_mask[bkr]*RT[(size_t)bkr*H2 + h];
  }
  PAI[idx] = v;
}

__global__ void k_sW(const float* ST, const float* Ws, float* SW){
  int idx = blockIdx.x*blockDim.x + threadIdx.x; if (idx >= BB*H2) return;
  int b = idx/H2, h2 = idx%H2;
  float acc = 0.f;
  for (int d = 0; d < H2; ++d) acc += ST[(size_t)b*H2+d]*Ws[(size_t)h2*H2+d];
  SW[idx] = acc;
}

__global__ void k_e(const float* SW, const float* PW, const float* Wqr_b, const float* Wc,
                    const float* Vr, const float* qr_cov, float* E){
  int idx = blockIdx.x*blockDim.x + threadIdx.x; if (idx >= BB*LLl) return;
  int b = idx/LLl;
  float cov = qr_cov[idx];
  const float* pw = PW + (size_t)idx*H2;
  const float* sw = SW + (size_t)b*H2;
  float acc = 0.f;
  for (int h2 = 0; h2 < H2; ++h2)
    acc += tanhf(sw[h2] + pw[h2] + Wqr_b[h2] + cov*Wc[h2]) * Vr[h2];
  E[idx] = acc;
}

__global__ void k_a(const float* E, const float* q_mask, const float* r_mask,
                    const float* qr_cov, float* A, float* out){
  int b = blockIdx.x, t = threadIdx.x;
  __shared__ float s[512];
  float val = (t < LLl) ? E[b*LLl + t] : -__builtin_inff();
  s[t] = val; __syncthreads();
  for (int off = 256; off > 0; off >>= 1) { if (t < off) s[t] = fmaxf(s[t], s[t+off]); __syncthreads(); }
  float m = s[0]; __syncthreads();
  float ex = (t < LLl) ? expf(val - m) : 0.f;
  s[t] = ex; __syncthreads();
  for (int off = 256; off > 0; off >>= 1) { if (t < off) s[t] += s[t+off]; __syncthreads(); }
  float tot = s[0]; __syncthreads();
  float mask = 0.f;
  if (t < LQq) mask = q_mask[b*LQq + t];
  else if (t < LLl) mask = r_mask[b*KKk*LRr + (t - LQq)];
  float av = (t < LLl) ? (ex/tot)*mask : 0.f;
  s[t] = av; __syncthreads();
  for (int off = 256; off > 0; off >>= 1) { if (t < off) s[t] += s[t+off]; __syncthreads(); }
  float tot2 = s[0];
  if (t < LLl) {
    float a = av/tot2;
    A[b*LLl + t] = a;
    out[OUT_AQR + b*LLl + t] = a;
    out[OUT_COV + b*LLl + t] = qr_cov[b*LLl + t] + a;
  }
}

__global__ void k_ct(const float* A, const float* PAI, float* CT, float* out){
  int idx = blockIdx.x*blockDim.x + threadIdx.x; if (idx >= BB*H2) return;
  int b = idx/H2, h = idx%H2;
  float acc = 0.f;
  for (int l = 0; l < LLl; ++l) acc += A[b*LLl+l]*PAI[((size_t)b*LLl+l)*H2 + h];
  CT[idx] = acc;
  out[OUT_CTT + idx] = acc;
}

__global__ void k_hst(const float* ST, const float* CT, float* HST){
  int idx = blockIdx.x*blockDim.x + threadIdx.x; if (idx >= BB*2*H2) return;
  int b = idx/(2*H2), j = idx%(2*H2);
  HST[idx] = (j < H2) ? ST[(size_t)b*H2+j] : CT[(size_t)b*H2 + (j-H2)];
}

__global__ void k_pgen(const float* HST, const float* pg_w, const float* pg_b, float* PG){
  int b = blockIdx.x, t = threadIdx.x;
  __shared__ float s[256];
  float acc = 0.f;
  for (int j = t; j < 2*H2; j += 256) acc += HST[(size_t)b*2*H2+j]*pg_w[j];
  s[t] = acc; __syncthreads();
  for (int off = 128; off > 0; off >>= 1) { if (t < off) s[t] += s[t+off]; __syncthreads(); }
  if (t == 0) PG[b] = sigm(s[0] + pg_b[0]);
}

__global__ void k_hid(const float* HST, const float* pv1_w, const float* pv1_b, float* HID){
  int idx = blockIdx.x*blockDim.x + threadIdx.x; if (idx >= BB*H2) return;
  int b = idx/H2, h2 = idx%H2;
  float acc = pv1_b[h2];
  const float* w = pv1_w + (size_t)h2*2*H2;
  for (int j = 0; j < 2*H2; ++j) acc += HST[(size_t)b*2*H2+j]*w[j];
  HID[idx] = acc;
}

__global__ void k_vstats(const float* LOG, float* VM, float* VS){
  int b = blockIdx.x, t = threadIdx.x;
  __shared__ float s[256];
  const float* row = LOG + (size_t)b*VOCB;
  float m = -__builtin_inff();
  for (int v = t; v < VOCB; v += 256) m = fmaxf(m, row[v]);
  s[t] = m; __syncthreads();
  for (int off = 128; off > 0; off >>= 1) { if (t < off) s[t] = fmaxf(s[t], s[t+off]); __syncthreads(); }
  float gm = s[0]; __syncthreads();
  float acc = 0.f;
  for (int v = t; v < VOCB; v += 256) acc += expf(row[v]-gm);
  s[t] = acc; __syncthreads();
  for (int off = 128; off > 0; off >>= 1) { if (t < off) s[t] += s[t+off]; __syncthreads(); }
  if (t == 0) { VM[b] = gm; VS[b] = s[0]; }
}

__global__ void k_final(const float* LOG, const float* VM, const float* VS, const float* PG, float* out){
  int idx = blockIdx.x*blockDim.x + threadIdx.x; if (idx >= BB*VOUT) return;
  int b = idx/VOUT, v = idx%VOUT;
  float r = 0.f;
  if (v < VOCB) r = PG[b]*expf(LOG[(size_t)b*VOCB+v]-VM[b])/VS[b];
  out[OUT_FD + idx] = r;
}

__global__ void k_scatter(const float* A, const float* PG, const int* bev, float* out){
  int idx = blockIdx.x*blockDim.x + threadIdx.x; if (idx >= BB*LLl) return;
  int b = idx/LLl;
  float add = (1.f - PG[b])*A[idx];
  atomicAdd(out + OUT_FD + (size_t)b*VOUT + bev[idx], add);
}

// ---------------- launcher ----------------
extern "C" void kernel_launch(void* const* d_in, const int* in_sizes, int n_in,
                              void* d_out, int out_size, void* d_ws, size_t ws_size,
                              hipStream_t stream) {
  const float* y_t   = (const float*)d_in[0];
  const float* h0    = (const float*)d_in[1];
  const float* c0    = (const float*)d_in[2];
  const float* c_t0  = (const float*)d_in[3];
  const float* o_m   = (const float*)d_in[4];
  const float* h_q   = (const float*)d_in[5];
  const float* q_mask= (const float*)d_in[6];
  const float* q_a   = (const float*)d_in[7];
  const float* h_r   = (const float*)d_in[8];
  const float* r_mask= (const float*)d_in[9];
  const float* r_a   = (const float*)d_in[10];
  const float* r_o   = (const float*)d_in[11];
  const float* qr_cov= (const float*)d_in[12];
  const float* xc_w  = (const float*)d_in[14];
  const float* xc_b  = (const float*)d_in[15];
  const float* w_ih  = (const float*)d_in[16];
  const float* w_hh  = (const float*)d_in[17];
  const float* b_ih  = (const float*)d_in[18];
  const float* b_hh  = (const float*)d_in[19];
  const float* dsq_w = (const float*)d_in[20];
  const float* dsa_w = (const float*)d_in[21];
  const float* dso_w = (const float*)d_in[22];
  const float* U_w   = (const float*)d_in[23];
  const float* Wc_w  = (const float*)d_in[24];
  const float* Ws_w  = (const float*)d_in[25];
  const float* Wqr_w = (const float*)d_in[26];
  const float* Wqr_b = (const float*)d_in[27];
  const float* Vr_w  = (const float*)d_in[28];
  const float* pg_w  = (const float*)d_in[29];
  const float* pg_b  = (const float*)d_in[30];
  const float* pv1_w = (const float*)d_in[31];
  const float* pv1_b = (const float*)d_in[32];
  const float* pv2_w = (const float*)d_in[33];
  const float* pv2_b = (const float*)d_in[34];
  const int*   bev   = (const int*)d_in[35];

  float* ws  = (float*)d_ws;
  float* out = (float*)d_out;

  float* ST  = ws + OFF_ST;  float* X   = ws + OFF_X;   float* TQ  = ws + OFF_TQ;
  float* WNQ = ws + OFF_WNQ; float* QAT = ws + OFF_QAT; float* TA  = ws + OFF_TA;
  float* TO  = ws + OFF_TO;  float* WNA = ws + OFF_WNA; float* WNO = ws + OFF_WNO;
  float* AQ  = ws + OFF_AQ;  float* AR  = ws + OFF_AR;  float* E   = ws + OFF_E;
  float* A   = ws + OFF_A;   float* CT  = ws + OFF_CT;  float* HST = ws + OFF_HST;
  float* PG  = ws + OFF_PG;  float* VM  = ws + OFF_VM;  float* VS  = ws + OFF_VS;
  float* HID = ws + OFF_HID; float* SW  = ws + OFF_SW;  float* QT  = ws + OFF_QT;
  float* QU  = ws + OFF_QU;  float* U   = ws + OFF_U;   float* RT  = ws + OFF_RT;
  float* PAI = ws + OFF_PAI; float* PW  = ws + OFF_PW;  float* LOG = ws + OFF_LOG;

  // LSTM step
  k_x   <<<8, 256, 0, stream>>>(y_t, c_t0, xc_w, xc_b, X);
  k_lstm<<<16, 256, 0, stream>>>(X, h0, c0, w_ih, w_hh, b_ih, b_hh, ST, out);
  // dynamic select q
  k_tq  <<<8, 256, 0, stream>>>(ST, dsq_w, TQ);
  k_wq  <<<BB, 32, 0, stream>>>(TQ, q_a, WNQ);
  k_qt  <<<1920, 256, 0, stream>>>(WNQ, h_q, QT);
  k_qat <<<8, 256, 0, stream>>>(WNQ, q_a, QAT);
  // dynamic select a / o over responses
  k_ta  <<<8, 256, 0, stream>>>(ST, QAT, dsa_w, TA);
  k_wr  <<<BB*KKk, 32, 0, stream>>>(TA, r_a, WNA);
  k_to  <<<8, 256, 0, stream>>>(ST, o_m, dso_w, TO);
  k_wr  <<<BB*KKk, 32, 0, stream>>>(TO, r_o, WNO);
  k_rt  <<<12800, 256, 0, stream>>>(WNA, WNO, h_r, RT);
  // co-attention
  k_gemm_qU<<<60*32, 32, 0, stream>>>(QT, U_w, QU);
  k_gemm_U <<<BB*KKk*28, 32, 0, stream>>>(QU, RT, U);
  k_alpha  <<<BB*KKk, 128, 0, stream>>>(U, AQ, AR);
  k_pai    <<<14720, 256, 0, stream>>>(AQ, AR, q_mask, r_mask, QT, RT, PAI);
  k_sW     <<<32, 256, 0, stream>>>(ST, Ws_w, SW);
  k_gemm_pW<<<460*32, 32, 0, stream>>>(PAI, Wqr_w, PW);
  k_e      <<<29, 256, 0, stream>>>(SW, PW, Wqr_b, Wc_w, Vr_w, qr_cov, E);
  k_a      <<<BB, 512, 0, stream>>>(E, q_mask, r_mask, qr_cov, A, out);
  k_ct     <<<32, 256, 0, stream>>>(A, PAI, CT, out);
  // output head
  k_hst    <<<64, 256, 0, stream>>>(ST, CT, HST);
  k_pgen   <<<BB, 256, 0, stream>>>(HST, pg_w, pg_b, PG);
  k_hid    <<<32, 256, 0, stream>>>(HST, pv1_w, pv1_b, HID);
  k_gemm_logits<<<VOCB/16, 32, 0, stream>>>(HID, pv2_w, pv2_b, LOG);
  k_vstats <<<BB, 256, 0, stream>>>(LOG, VM, VS);
  k_final  <<<3132, 256, 0, stream>>>(LOG, VM, VS, PG, out);
  k_scatter<<<29, 256, 0, stream>>>(A, PG, bev, out);
}
